// KiperwasserDependencyParser_47588237640459
// MI455X (gfx1250) — matrix operand order repaired and checked by reading, not compiled, not run on hardware
//
#include <hip/hip_runtime.h>
#include <hip/hip_bf16.h>

typedef __bf16 bf16_t;
typedef __attribute__((ext_vector_type(16))) __bf16 v16bf;
typedef __attribute__((ext_vector_type(8)))  __bf16 v8bf;
typedef __attribute__((ext_vector_type(8)))  float  v8f;

#define SEQ   1024
#define EHN   512
#define G4    2048   // 4*EH
#define HIDP  128    // padded input dim (125 -> 128)
#define MLPH  256
#define HCAT  1024   // 2*EH

__device__ __forceinline__ float sigmoidf_(float x){ return 1.0f/(1.0f+__expf(-x)); }

// ---------------- prep kernels ----------------

__global__ void k_embed_bf16(const int* __restrict__ wi, const int* __restrict__ pi,
                             const float* __restrict__ wemb, const float* __restrict__ pemb,
                             bf16_t* __restrict__ out){
  int idx = blockIdx.x*blockDim.x + threadIdx.x;
  if (idx >= SEQ*HIDP) return;
  int i = idx >> 7, c = idx & 127;
  float v = 0.0f;
  if (c < 100)      v = wemb[(size_t)wi[i]*100 + c];
  else if (c < 125) v = pemb[(size_t)pi[i]*25 + (c-100)];
  out[idx] = (bf16_t)v;
}

__global__ void k_pad_bf16(bf16_t* __restrict__ dst, const float* __restrict__ src,
                           int rows, int srcK, int dstK){
  int idx = blockIdx.x*blockDim.x + threadIdx.x;
  if (idx >= rows*dstK) return;
  int r = idx / dstK, c = idx % dstK;
  dst[idx] = (bf16_t)((c < srcK) ? src[(size_t)r*srcK + c] : 0.0f);
}

__global__ void k_addvec(float* __restrict__ dst, const float* __restrict__ a,
                         const float* __restrict__ b, int n){
  int idx = blockIdx.x*blockDim.x + threadIdx.x;
  if (idx < n) dst[idx] = a[idx] + b[idx];
}

__global__ void k_f2bf(bf16_t* __restrict__ dst, const float* __restrict__ src, int n){
  int idx = blockIdx.x*blockDim.x + threadIdx.x;
  if (idx < n) dst[idx] = (bf16_t)src[idx];
}

// split W1 (256 x 2048) into head (first 1024 cols) and dep (last 1024 cols), bf16
__global__ void k_w1split(bf16_t* __restrict__ wh, bf16_t* __restrict__ wd,
                          const float* __restrict__ W1){
  int idx = blockIdx.x*blockDim.x + threadIdx.x;
  if (idx >= MLPH*HCAT) return;
  int r = idx / HCAT, c = idx % HCAT;
  wh[idx] = (bf16_t)W1[(size_t)r*(2*HCAT) + c];
  wd[idx] = (bf16_t)W1[(size_t)r*(2*HCAT) + HCAT + c];
}

// ---------------- WMMA GEMM: C[M,N] = A[M,K](bf16) @ W[N,K]^T(bf16) + bias[N] ----------------
// block = 256 threads = 8 waves; each wave owns a 16 x (16*NBLK) C strip:
// one A fragment feeds NBLK back-to-back v_wmma ops per K step (XDL-friendly).
// grid = (N/(128*NBLK), M/16). K % 32 == 0.
template<int NBLK>
__global__ void k_gemm_bf16(float* __restrict__ C, const bf16_t* __restrict__ A,
                            const bf16_t* __restrict__ W, const float* __restrict__ bias,
                            int M, int N, int K){
  const int wv     = threadIdx.x >> 5;
  const int lane   = threadIdx.x & 31;
  const int tileN0 = (blockIdx.x*8 + wv) * (16*NBLK);
  const int tileM  = blockIdx.y << 4;
  if (tileN0 >= N || tileM >= M) return;
  const int lhi = lane >> 4;   // half-wave select
  const int l15 = lane & 15;

  const bf16_t* arow = A + (size_t)(tileM + l15)*K;   // A-frag: M = l15 (both halves)
  const bf16_t* brow[NBLK];
  #pragma unroll
  for (int nb = 0; nb < NBLK; ++nb)
    brow[nb] = W + (size_t)(tileN0 + nb*16 + l15)*K;  // B-frag: N = l15 (both halves)

  v8f acc[NBLK];
  #pragma unroll
  for (int nb = 0; nb < NBLK; ++nb) acc[nb] = (v8f){};

  for (int k = 0; k < K; k += 32){
    // A 16x32 bf16 fragment: lanes 0-15 elems = K[k..k+15];
    // lanes 16-31 elems = K[k+16..k+31]
    v8bf a0 = *(const v8bf*)(arow + k + lhi*16);
    v8bf a1 = *(const v8bf*)(arow + k + lhi*16 + 8);
    v16bf af;
    #pragma unroll
    for (int e = 0; e < 8; ++e){ af[e] = a0[e]; af[e+8] = a1[e]; }
    if (k + 64 < K){
      __builtin_prefetch(arow + k + 64, 0, 0);
      __builtin_prefetch(brow[0] + k + 64, 0, 0);
    }
    // B 32x16 fragment: lane n holds K[k..k+15] (lanes 0-15) / K[k+16..k+31] (lanes 16-31)
    #pragma unroll
    for (int nb = 0; nb < NBLK; ++nb){
      v16bf bfv = *(const v16bf*)(brow[nb] + k + lhi*16);
      acc[nb] = __builtin_amdgcn_wmma_f32_16x16x32_bf16(false, af, false, bfv,
                                                        (short)0, acc[nb], false, false);
    }
  }
  #pragma unroll
  for (int nb = 0; nb < NBLK; ++nb){
    const int col = tileN0 + nb*16 + l15;
    const float bv = bias ? bias[col] : 0.0f;
    #pragma unroll
    for (int r = 0; r < 8; ++r){
      int row = tileM + r + lhi*8;   // C/D layout: lanes 16-31 hold M = r+8
      C[(size_t)row*N + col] = acc[nb][r] + bv;
    }
  }
}

// ---------------- persistent BiLSTM recurrence ----------------
// grid = 2 (dir), block = 1024. h in LDS; Whh streamed from L2 each step.
// thread t computes gate rows t and t+1024; threads <512 own hidden unit t (c in reg).
__global__ void k_lstm(const float* __restrict__ Xf, const float* __restrict__ Xb,
                       const float* __restrict__ WhhF, const float* __restrict__ WhhB,
                       float* __restrict__ H){
  __shared__ float h_s[EHN];
  __shared__ float g_s[G4];
  const int dir = blockIdx.x;
  const float* __restrict__ X   = dir ? Xb   : Xf;
  const float* __restrict__ Whh = dir ? WhhB : WhhF;
  const int t = threadIdx.x;
  float cst = 0.0f;
  if (t < EHN) h_s[t] = 0.0f;
  __syncthreads();
  const float* w0 = Whh + (size_t)t*EHN;
  const float* w1 = Whh + (size_t)(t + 1024)*EHN;
  for (int s = 0; s < SEQ; ++s){
    const int time = dir ? (SEQ-1-s) : s;
    float acc0 = X[(size_t)time*G4 + t];
    float acc1 = X[(size_t)time*G4 + t + 1024];
    #pragma unroll 4
    for (int j = 0; j < EHN; j += 4){
      float4 hv = *(const float4*)(h_s + j);
      float4 a4 = *(const float4*)(w0 + j);
      float4 b4 = *(const float4*)(w1 + j);
      acc0 += a4.x*hv.x + a4.y*hv.y + a4.z*hv.z + a4.w*hv.w;
      acc1 += b4.x*hv.x + b4.y*hv.y + b4.z*hv.z + b4.w*hv.w;
    }
    g_s[t]        = acc0;
    g_s[t + 1024] = acc1;
    __syncthreads();
    if (t < EHN){
      float ig = sigmoidf_(g_s[t]);
      float fg = sigmoidf_(g_s[EHN + t]);
      float gg = tanhf    (g_s[2*EHN + t]);
      float og = sigmoidf_(g_s[3*EHN + t]);
      cst = fg*cst + ig*gg;
      float hv = og * tanhf(cst);
      h_s[t] = hv;
      H[(size_t)time*HCAT + dir*EHN + t] = hv;
    }
    __syncthreads();
  }
}

// ---------------- pair scorer: scores[i,j] = sum_k tanh(a[i,k]+b[j,k]) * W2[k] ----------------
// (b1 folded into a via GEMM bias; b2 cancels in log_softmax)
// grid = SEQ (i), block = 256 = 8 waves; each wave sweeps j with stride 8.
__global__ void k_pair(const float* __restrict__ A, const float* __restrict__ B,
                       const float* __restrict__ W2, float* __restrict__ S){
  __shared__ float aS[MLPH];
  __shared__ float w2S[MLPH];
  const int i = blockIdx.x, tid = threadIdx.x;
  aS[tid]  = A[(size_t)i*MLPH + tid];
  w2S[tid] = W2[tid];
  __syncthreads();
  const int wv = tid >> 5, lane = tid & 31;
  for (int j = wv; j < SEQ; j += 8){
    const float* bj = B + (size_t)j*MLPH;
    float p = 0.0f;
    #pragma unroll
    for (int k = lane; k < MLPH; k += 32)
      p += tanhf(aS[k] + bj[k]) * w2S[k];
    #pragma unroll
    for (int off = 16; off; off >>= 1) p += __shfl_xor(p, off, 32);
    if (lane == 0) S[(size_t)i*SEQ + j] = p;
  }
}

// ---------------- per-column logsumexp + masked pick ----------------
__global__ void k_colloss(const float* __restrict__ S, const int* __restrict__ heads,
                          float* __restrict__ colv){
  __shared__ float red[256];
  const int j = blockIdx.x, t = threadIdx.x;
  float m = -3.4e38f;
  for (int i = t; i < SEQ; i += 256) m = fmaxf(m, S[(size_t)i*SEQ + j]);
  red[t] = m; __syncthreads();
  for (int o = 128; o; o >>= 1){ if (t < o) red[t] = fmaxf(red[t], red[t+o]); __syncthreads(); }
  m = red[0]; __syncthreads();
  float s = 0.0f;
  for (int i = t; i < SEQ; i += 256) s += __expf(S[(size_t)i*SEQ + j] - m);
  red[t] = s; __syncthreads();
  for (int o = 128; o; o >>= 1){ if (t < o) red[t] += red[t+o]; __syncthreads(); }
  if (t == 0){
    float lse = m + __logf(red[0]);
    int h = heads[j];
    colv[j] = (h >= 0) ? (S[(size_t)h*SEQ + j] - lse) : 0.0f;
  }
}

__global__ void k_final(const float* __restrict__ colv, float* __restrict__ out){
  __shared__ float red[256];
  const int t = threadIdx.x;
  red[t] = colv[t] + colv[t+256] + colv[t+512] + colv[t+768];
  __syncthreads();
  for (int o = 128; o; o >>= 1){ if (t < o) red[t] += red[t+o]; __syncthreads(); }
  if (t == 0) out[0] = -red[0] / (float)SEQ;
}

// ---------------- launch ----------------
extern "C" void kernel_launch(void* const* d_in, const int* in_sizes, int n_in,
                              void* d_out, int out_size, void* d_ws, size_t ws_size,
                              hipStream_t stream){
  (void)in_sizes; (void)n_in; (void)out_size; (void)ws_size;
  const int*   word_idx = (const int*)  d_in[0];
  const int*   pos_idx  = (const int*)  d_in[1];
  const int*   heads    = (const int*)  d_in[2];
  const float* wemb     = (const float*)d_in[3];
  const float* pemb     = (const float*)d_in[4];
  const float* Wih_f    = (const float*)d_in[5];
  const float* Whh_f    = (const float*)d_in[6];
  const float* bih_f    = (const float*)d_in[7];
  const float* bhh_f    = (const float*)d_in[8];
  const float* Wih_b    = (const float*)d_in[9];
  const float* Whh_b    = (const float*)d_in[10];
  const float* bih_b    = (const float*)d_in[11];
  const float* bhh_b    = (const float*)d_in[12];
  const float* W1       = (const float*)d_in[13];
  const float* b1       = (const float*)d_in[14];
  const float* W2       = (const float*)d_in[15];
  // d_in[16] = b2: constant per column, cancels inside log_softmax -> unused.

  char* ws = (char*)d_ws;
  size_t off = 0;
  auto take = [&](size_t bytes)->char*{
    char* p = ws + off; off = (off + bytes + 255) & ~(size_t)255; return p;
  };
  bf16_t* Ebf  = (bf16_t*)take((size_t)SEQ*HIDP*2);
  bf16_t* WfB  = (bf16_t*)take((size_t)G4*HIDP*2);
  bf16_t* WbB  = (bf16_t*)take((size_t)G4*HIDP*2);
  float*  bsf  = (float*) take((size_t)G4*4);
  float*  bsb  = (float*) take((size_t)G4*4);
  bf16_t* W1hB = (bf16_t*)take((size_t)MLPH*HCAT*2);
  bf16_t* W1dB = (bf16_t*)take((size_t)MLPH*HCAT*2);
  float*  Xf   = (float*) take((size_t)SEQ*G4*4);
  float*  Xb   = (float*) take((size_t)SEQ*G4*4);
  float*  H    = (float*) take((size_t)SEQ*HCAT*4);
  bf16_t* Hbf  = (bf16_t*)take((size_t)SEQ*HCAT*2);
  float*  Aout = (float*) take((size_t)SEQ*MLPH*4);
  float*  Bout = (float*) take((size_t)SEQ*MLPH*4);
  float*  Sc   = (float*) take((size_t)SEQ*SEQ*4);
  float*  colv = (float*) take((size_t)SEQ*4);

  k_embed_bf16<<<(SEQ*HIDP+255)/256, 256, 0, stream>>>(word_idx, pos_idx, wemb, pemb, Ebf);
  k_pad_bf16<<<(G4*HIDP+255)/256, 256, 0, stream>>>(WfB, Wih_f, G4, 125, HIDP);
  k_pad_bf16<<<(G4*HIDP+255)/256, 256, 0, stream>>>(WbB, Wih_b, G4, 125, HIDP);
  k_addvec<<<(G4+255)/256, 256, 0, stream>>>(bsf, bih_f, bhh_f, G4);
  k_addvec<<<(G4+255)/256, 256, 0, stream>>>(bsb, bih_b, bhh_b, G4);
  k_w1split<<<(MLPH*HCAT+255)/256, 256, 0, stream>>>(W1hB, W1dB, W1);

  // input projections: N=2048, NBLK=4 -> block covers 512 cols, grid.x = 4
  dim3 g1(G4/(128*4), SEQ/16);
  k_gemm_bf16<4><<<g1, 256, 0, stream>>>(Xf, Ebf, WfB, bsf, SEQ, G4, HIDP);
  k_gemm_bf16<4><<<g1, 256, 0, stream>>>(Xb, Ebf, WbB, bsb, SEQ, G4, HIDP);

  k_lstm<<<2, 1024, 0, stream>>>(Xf, Xb, Whh_f, Whh_b, H);
  k_f2bf<<<(SEQ*HCAT+255)/256, 256, 0, stream>>>(Hbf, H, SEQ*HCAT);

  // head/dep projections: N=256, NBLK=2 -> block covers 256 cols, grid.x = 1
  dim3 g2(MLPH/(128*2), SEQ/16);
  k_gemm_bf16<2><<<g2, 256, 0, stream>>>(Aout, Hbf, W1hB, b1,      SEQ, MLPH, HCAT);
  k_gemm_bf16<2><<<g2, 256, 0, stream>>>(Bout, Hbf, W1dB, nullptr, SEQ, MLPH, HCAT);

  k_pair<<<SEQ, 256, 0, stream>>>(Aout, Bout, W2, Sc);
  k_colloss<<<SEQ, 256, 0, stream>>>(Sc, heads, colv);
  k_final<<<1, 256, 0, stream>>>(colv, (float*)d_out);
}
